// WaveletConv2D_52759378264381
// MI455X (gfx1250) — matrix operand order repaired and checked by reading, not compiled, not run on hardware
//
#include <hip/hip_runtime.h>
#include <hip/hip_bf16.h>
#include <stdint.h>

// ---------------------------------------------------------------------------
// Wavelet filter taps (Daubechies-8 analysis h0; others derived:
//  h1[k] = h0[7-k] * (k even ? -1 : +1)
//  g0[k] = h0[7-k]
//  g1[k] = h1[7-k] = h0[k] * (k odd ? -1 : +1))
// ---------------------------------------------------------------------------
__constant__ float c_h0[8] = {0.0322231f, -0.01260397f, -0.09921954f, 0.2978578f,
                              0.80373875f, 0.49761867f, -0.02963553f, -0.07576571f};

__device__ __forceinline__ float f_h0(int k) { return c_h0[k]; }
__device__ __forceinline__ float f_h1(int k) { return c_h0[7 - k] * ((k & 1) ? 1.f : -1.f); }
__device__ __forceinline__ float f_g0(int k) { return c_h0[7 - k]; }
__device__ __forceinline__ float f_g1(int k) { return c_h0[k] * ((k & 1) ? -1.f : 1.f); }

#define BATCH 16
#define CH 64

// ---------------------------------------------------------------------------
// small float4 helpers (avoid depending on HIP vector operator overloads)
// ---------------------------------------------------------------------------
struct F4 { float x, y, z, w; };
__device__ __forceinline__ F4 ld4(const float* p) { return *reinterpret_cast<const F4*>(p); }
__device__ __forceinline__ void st4(float* p, F4 v) { *reinterpret_cast<F4*>(p) = v; }
__device__ __forceinline__ F4 fma4(F4 a, float s, F4 b) {
  b.x = fmaf(a.x, s, b.x); b.y = fmaf(a.y, s, b.y);
  b.z = fmaf(a.z, s, b.z); b.w = fmaf(a.w, s, b.w);
  return b;
}

// ---------------------------------------------------------------------------
// Analysis: depthwise 8x8 stride-2 SAME conv (pad lo=3), 4 subbands at once,
// separable row/col factoring. Thread = (b, oh, ow, c4) with 4 channels.
// ---------------------------------------------------------------------------
__global__ void __launch_bounds__(256)
wavelet_analysis(const float* __restrict__ in,
                 float* __restrict__ oLL, float* __restrict__ oLH,
                 float* __restrict__ oHL, float* __restrict__ oHH,
                 int Hin) {
  const int Ho = Hin >> 1;
  int idx = blockIdx.x * blockDim.x + threadIdx.x;
  int c4 = idx & 15; int rest = idx >> 4;
  int ow = rest % Ho; rest /= Ho;
  int oh = rest % Ho; int b = rest / Ho;
  if (b >= BATCH) return;

  F4 LL{}, LH{}, HL{}, HH{};
  for (int kh = 0; kh < 8; ++kh) {
    int ih = 2 * oh + kh - 3;
    if (ih < 0 || ih >= Hin) continue;
    F4 t0{}, t1{};
    const float* row = in + (((size_t)b * Hin + ih) * Hin) * CH + c4 * 4;
    for (int kw = 0; kw < 8; ++kw) {
      int iw = 2 * ow + kw - 3;
      if (iw < 0 || iw >= Hin) continue;
      F4 v = ld4(row + (size_t)iw * CH);
      t0 = fma4(v, f_h0(kw), t0);
      t1 = fma4(v, f_h1(kw), t1);
    }
    LL = fma4(t0, f_h0(kh), LL);
    LH = fma4(t1, f_h0(kh), LH);
    HL = fma4(t0, f_h1(kh), HL);
    HH = fma4(t1, f_h1(kh), HH);
  }
  size_t o = (((size_t)b * Ho + oh) * Ho + ow) * CH + c4 * 4;
  st4(oLL + o, LL); st4(oLH + o, LH); st4(oHL + o, HL); st4(oHH + o, HH);
}

// ---------------------------------------------------------------------------
// Synthesis: out[oh,ow] = sum of 4 subbands, upsample-by-2 then 8x8 stride-1
// SAME conv (pad lo=3). Only taps hitting even upsampled coords contribute.
// ---------------------------------------------------------------------------
__global__ void __launch_bounds__(256)
wavelet_synthesis(const float* __restrict__ yA, const float* __restrict__ yLH,
                  const float* __restrict__ yHL, const float* __restrict__ yHH,
                  float* __restrict__ out, int S) {
  const int Ho = S * 2;
  int idx = blockIdx.x * blockDim.x + threadIdx.x;
  int c4 = idx & 15; int rest = idx >> 4;
  int ow = rest % Ho; rest /= Ho;
  int oh = rest % Ho; int b = rest / Ho;
  if (b >= BATCH) return;

  F4 acc{};
  for (int kh = 0; kh < 8; ++kh) {
    int ih = oh + kh - 3;
    if (ih < 0 || ih >= Ho || (ih & 1)) continue;
    int mh = ih >> 1;
    F4 sA{}, sLH{}, sHL{}, sHH{};
    size_t rb = (((size_t)b * S + mh) * S) * CH + c4 * 4;
    for (int kw = 0; kw < 8; ++kw) {
      int iw = ow + kw - 3;
      if (iw < 0 || iw >= Ho || (iw & 1)) continue;
      size_t off = rb + (size_t)(iw >> 1) * CH;
      float g0w = f_g0(kw), g1w = f_g1(kw);
      sA  = fma4(ld4(yA  + off), g0w, sA);
      sLH = fma4(ld4(yLH + off), g1w, sLH);
      sHL = fma4(ld4(yHL + off), g0w, sHL);
      sHH = fma4(ld4(yHH + off), g1w, sHH);
    }
    float gh0 = f_g0(kh), gh1 = f_g1(kh);
    acc = fma4(sA,  gh0, acc);
    acc = fma4(sLH, gh0, acc);
    acc = fma4(sHL, gh1, acc);
    acc = fma4(sHH, gh1, acc);
  }
  st4(out + (((size_t)b * Ho + oh) * Ho + ow) * CH + c4 * 4, acc);
}

// ---------------------------------------------------------------------------
// Position-dependent channel-mix via WMMA:
//   out[b,h,w,o] = sum_i act[b,h,w,i] * wgt[i,o,w,h]     (in place over act)
// Workgroup = 256 threads (8 waves), owns 16 positions: w = blockIdx.x,
// h-chunk = blockIdx.y (2D grid avoids integer division in the preamble).
// Weights staged into LDS as bf16 WMMA B-fragments with coalesced global
// reads (consecutive lanes walk h). Per position: A=16x64 (batch x ch),
// two chained v_wmma_f32_16x16x32_bf16 per 16-wide N tile, f32 accumulate.
// ---------------------------------------------------------------------------
typedef __attribute__((ext_vector_type(16))) __bf16 v16bf;
typedef __attribute__((ext_vector_type(8)))  float  v8f;

union Frag16 { v16bf v; unsigned short u[16]; };

__device__ __forceinline__ unsigned short f2bf(float f) {  // round-to-nearest-even
  unsigned int u = __float_as_uint(f);
  return (unsigned short)((u + 0x7FFFu + ((u >> 16) & 1u)) >> 16);
}

#define POS_STRIDE 4104   // 8 fragments * 512 bf16 + 8 pad (LDS bank spread)

__global__ void __launch_bounds__(256)
local_conv_wmma(float* act, const float* __restrict__ wgt, int Hs, int Ws) {
  extern __shared__ unsigned short lds[];
  const int tid = threadIdx.x;
  const int w   = blockIdx.x;        // position column
  const int hc  = blockIdx.y;        // h-chunk of 16 positions

  // ---- stage 16 positions' 64x64 weight matrices into LDS (fragment order)
#pragma unroll 4
  for (int f = tid; f < 16 * 4096; f += 256) {
    int pl = f & 15;                 // consecutive lanes -> consecutive h: coalesced
    int io = f >> 4;
    int i = io >> 6, o = io & 63;
    int h = hc * 16 + pl;
    float wv = wgt[(((size_t)i * 64 + o) * Ws + w) * Hs + h];
    // B-fragment layout (32x16 bf16): lanes 0-15 hold K=0..15 (elem e=K),
    // lanes 16-31 hold K=16..31; col n = lane & 15.
    int nt = o >> 4, n = o & 15;
    int kb = i >> 5, kl = i & 31;
    int half = kl >> 4, e = kl & 15;
    lds[pl * POS_STRIDE + (nt * 2 + kb) * 512 + (n + 16 * half) * 16 + e] = f2bf(wv);
  }
  __syncthreads();

  const int wave = tid >> 5, lane = tid & 31;
  const int bRow = lane & 15, halfL = lane >> 4;

  for (int pp = 0; pp < 2; ++pp) {
    int pl = wave * 2 + pp;
    int h = hc * 16 + pl;
    const float* actp = act + (((size_t)bRow * Hs + h) * Ws + w) * CH;

    // A fragments (16x32 bf16 each): lane row M = bRow; the lane's 16 elems
    // are K = 16*half + e (matching the B staging), i.e. one contiguous
    // 16-float run in channel space.
    Frag16 a[2];
#pragma unroll
    for (int kb = 0; kb < 2; ++kb) {
      int i0 = kb * 32 + 16 * halfL;
      int i1 = i0 + 8;
      F4 p0 = ld4(actp + i0), p1 = ld4(actp + i0 + 4);
      F4 p2 = ld4(actp + i1), p3 = ld4(actp + i1 + 4);
      a[kb].u[0]  = f2bf(p0.x); a[kb].u[1]  = f2bf(p0.y);
      a[kb].u[2]  = f2bf(p0.z); a[kb].u[3]  = f2bf(p0.w);
      a[kb].u[4]  = f2bf(p1.x); a[kb].u[5]  = f2bf(p1.y);
      a[kb].u[6]  = f2bf(p1.z); a[kb].u[7]  = f2bf(p1.w);
      a[kb].u[8]  = f2bf(p2.x); a[kb].u[9]  = f2bf(p2.y);
      a[kb].u[10] = f2bf(p2.z); a[kb].u[11] = f2bf(p2.w);
      a[kb].u[12] = f2bf(p3.x); a[kb].u[13] = f2bf(p3.y);
      a[kb].u[14] = f2bf(p3.z); a[kb].u[15] = f2bf(p3.w);
    }

    const unsigned short* bbase = &lds[pl * POS_STRIDE + lane * 16];
#pragma unroll
    for (int nt = 0; nt < 4; ++nt) {
      Frag16 b0, b1;
#pragma unroll
      for (int e = 0; e < 16; ++e) {
        b0.u[e] = bbase[(nt * 2 + 0) * 512 + e];
        b1.u[e] = bbase[(nt * 2 + 1) * 512 + e];
      }
      v8f c = {0.f, 0.f, 0.f, 0.f, 0.f, 0.f, 0.f, 0.f};
      c = __builtin_amdgcn_wmma_f32_16x16x32_bf16(false, a[0].v, false, b0.v, (short)0, c, false, false);
      c = __builtin_amdgcn_wmma_f32_16x16x32_bf16(false, a[1].v, false, b1.v, (short)0, c, false, false);
      // C layout: col = lane&15, rows M = r + 8*(lane>>4). In-place store is
      // safe: this workgroup is the only reader/writer of its positions.
      int oc = nt * 16 + bRow;
#pragma unroll
      for (int r = 0; r < 8; ++r) {
        int bb = r + 8 * halfL;
        act[(((size_t)bb * Hs + h) * Ws + w) * CH + oc] = c[r];
      }
    }
  }
}

// ---------------------------------------------------------------------------
extern "C" void kernel_launch(void* const* d_in, const int* in_sizes, int n_in,
                              void* d_out, int out_size, void* d_ws, size_t ws_size,
                              hipStream_t stream) {
  (void)in_sizes; (void)n_in; (void)out_size; (void)ws_size;
  const float* x    = (const float*)d_in[0];
  const float* wLL  = (const float*)d_in[1];
  const float* wLH0 = (const float*)d_in[2];
  const float* wHL0 = (const float*)d_in[3];
  const float* wHH0 = (const float*)d_in[4];
  const float* wLH1 = (const float*)d_in[5];
  const float* wHL1 = (const float*)d_in[6];
  const float* wHH1 = (const float*)d_in[7];
  float* out = (float*)d_out;
  float* ws  = (float*)d_ws;

  const size_t n0 = (size_t)16 * 64 * 64 * 64;  // level-0 subband elems
  const size_t n1 = (size_t)16 * 32 * 32 * 64;  // level-1 subband elems
  float* A0 = ws;            // cLH0 -> yLH0 (in place)
  float* A1 = A0 + n0;       // cHL0 -> yHL0
  float* A2 = A1 + n0;       // cHH0 -> yHH0
  float* A3 = A2 + n0;       // cLL0, later reused for r1 (level-1 recon)
  float* B0 = A3 + n0;       // cLL1 -> yA
  float* B1 = B0 + n1;       // cLH1 -> yLH1
  float* B2 = B1 + n1;       // cHL1 -> yHL1
  float* B3 = B2 + n1;       // cHH1 -> yHH1

  // Analysis
  wavelet_analysis<<<(16 * 64 * 64 * 16) / 256, 256, 0, stream>>>(x,  A3, A0, A1, A2, 128);
  wavelet_analysis<<<(16 * 32 * 32 * 16) / 256, 256, 0, stream>>>(A3, B0, B1, B2, B3, 64);

  // Position-dependent channel mixes (WMMA, in place)
  const size_t ldsb = (size_t)16 * POS_STRIDE * sizeof(unsigned short);  // 131328 B
  hipFuncSetAttribute((const void*)local_conv_wmma,
                      hipFuncAttributeMaxDynamicSharedMemorySize, (int)ldsb);
  local_conv_wmma<<<dim3(32, 32 / 16), 256, ldsb, stream>>>(B0, wLL,  32, 32);
  local_conv_wmma<<<dim3(64, 64 / 16), 256, ldsb, stream>>>(A0, wLH0, 64, 64);
  local_conv_wmma<<<dim3(64, 64 / 16), 256, ldsb, stream>>>(A1, wHL0, 64, 64);
  local_conv_wmma<<<dim3(64, 64 / 16), 256, ldsb, stream>>>(A2, wHH0, 64, 64);
  local_conv_wmma<<<dim3(32, 32 / 16), 256, ldsb, stream>>>(B1, wLH1, 32, 32);
  local_conv_wmma<<<dim3(32, 32 / 16), 256, ldsb, stream>>>(B2, wHL1, 32, 32);
  local_conv_wmma<<<dim3(32, 32 / 16), 256, ldsb, stream>>>(B3, wHH1, 32, 32);

  // Synthesis: level 1 -> r1 (into A3, free after second analysis), level 0 -> out
  wavelet_synthesis<<<(16 * 64 * 64 * 16) / 256,   256, 0, stream>>>(B0, B1, B2, B3, A3, 32);
  wavelet_synthesis<<<(16 * 128 * 128 * 16) / 256, 256, 0, stream>>>(A3, A0, A1, A2, out, 64);
}
